// PoseRefiner_21010980012576
// MI455X (gfx1250) — compile-verified, matrix operand order checked by reading, not run on hardware
//
#include <hip/hip_runtime.h>
#include <hip/hip_bf16.h>

// ---------------- problem constants ----------------
#define BB   2
#define TT   64
#define NN   543
#define CC   256
#define HH2  128
#define CO   3
#define KT   9      // temporal kernel taps
#define LY   9      // layers
#define SKK  5      // smoothing taps
#define NP   544    // padded node count (34 * 16)
#define MROWS (BB*TT*NN)          // 69504
#define BT    (BB*TT)             // 128
#define BN_EPS 1e-5f

typedef __bf16 bf16;
typedef __attribute__((ext_vector_type(16))) __bf16 v16bf;
typedef __attribute__((ext_vector_type(8)))  __bf16 bf16x8;
typedef __attribute__((ext_vector_type(8)))  float  v8f;
typedef __attribute__((ext_vector_type(4)))  unsigned int u32x4;
typedef __attribute__((ext_vector_type(4)))  int i32x4;
typedef __attribute__((ext_vector_type(8)))  int i32x8;

static __device__ __forceinline__ v8f wmma_bf16(v16bf a, v16bf b, v8f c) {
  // (neg_a, A, neg_b, B, c_mod, C, reuse_a, reuse_b)
  return __builtin_amdgcn_wmma_f32_16x16x32_bf16(false, a, false, b, (short)0, c, false, false);
}

// Build a 16-element bf16 fragment from two contiguous 8-element (16B) runs.
static __device__ __forceinline__ v16bf load_frag(const bf16* p0, const bf16* p1) {
  v16bf r;
  ((bf16x8*)&r)[0] = *(const bf16x8*)p0;
  ((bf16x8*)&r)[1] = *(const bf16x8*)p1;
  return r;
}

// ---------------- weight prep ----------------
__global__ __launch_bounds__(256) void k_prep_gcnw(const float* gw, bf16* gwt) {
  int idx = blockIdx.x * 256 + threadIdx.x;
  if (idx >= LY * CC * CC) return;
  int i  = idx / (CC * CC);
  int r  = idx % (CC * CC);
  int co = r / CC, ci = r % CC;
  gwt[idx] = (bf16)gw[(size_t)i * CC * CC + (size_t)ci * CC + co]; // [i][co][ci]
}

__global__ __launch_bounds__(256) void k_prep_tcnw(const float* tw, bf16* twt) {
  int idx = blockIdx.x * 256 + threadIdx.x;
  if (idx >= LY * KT * CC * CC) return;
  int i  = idx / (KT * CC * CC);
  int r  = idx % (KT * CC * CC);
  int k  = r / (CC * CC);
  int r2 = r % (CC * CC);
  int co = r2 / CC, ci = r2 % CC;
  // src: [i][co][ci][k] -> dst: [i][k][co][ci]
  twt[idx] = (bf16)tw[(((size_t)i * CC + co) * CC + ci) * KT + k];
}

__global__ __launch_bounds__(256) void k_prep_adj(const float* adj, bf16* adjp) {
  int idx = blockIdx.x * 256 + threadIdx.x;
  if (idx >= NP * NP) return;
  int r = idx / NP, c = idx % NP;
  adjp[idx] = (r < NN && c < NN) ? (bf16)adj[(size_t)r * NN + c] : (bf16)0.0f;
}

__global__ __launch_bounds__(256) void k_prep_w1(const float* w1, bf16* w1t) {
  int idx = blockIdx.x * 256 + threadIdx.x;
  if (idx >= HH2 * CC) return;
  int h = idx / CC, c = idx % CC;
  w1t[idx] = (bf16)w1[(size_t)c * HH2 + h]; // [h][c]
}

__global__ __launch_bounds__(256) void k_prep_bn(const float* g, const float* b,
                                                 const float* m, const float* v,
                                                 float* scale, float* shift) {
  int idx = blockIdx.x * 256 + threadIdx.x;
  if (idx >= LY * CC) return;
  float s = g[idx] * rsqrtf(v[idx] + BN_EPS);
  scale[idx] = s;
  shift[idx] = b[idx] - m[idx] * s;
}

// ---------------- input embedding: x = poses @ w_in + b_in ----------------
__global__ __launch_bounds__(256) void k_embed(const float* poses, const float* w_in,
                                               const float* b_in, float* xf, bf16* xb) {
  int idx = blockIdx.x * 256 + threadIdx.x; // row*C + c, grid sized exactly
  int row = idx / CC, c = idx % CC;
  float s = b_in[c];
  const float* p = poses + (size_t)row * CO;
  s += p[0] * w_in[0 * CC + c];
  s += p[1] * w_in[1 * CC + c];
  s += p[2] * w_in[2 * CC + c];
  xf[idx] = s;
  xb[idx] = (bf16)s;
}

// ---------------- z = x @ W (WMMA, 16x64 per wave) ----------------
// A: x_bf [M][256] row-major; Bt: Wt [256 cout][256 cin]; Z: bf16 [M][256]
__global__ __launch_bounds__(256) void k_gemm_xw(const bf16* A, const bf16* Bt, bf16* Z) {
  int w = threadIdx.x >> 5, lane = threadIdx.x & 31;
  int wt = blockIdx.x * 8 + w;                // (M/16)*4 wave-tiles, exact
  int mt = wt >> 2, ng = wt & 3;              // ng: 64-wide n-group
  int l16 = lane & 15, h8 = (lane >> 4) * 8, hh = lane >> 4;
  const bf16* arow = A  + (size_t)(mt * 16 + l16) * CC;
  const bf16* brow = Bt + (size_t)(ng * 64 + l16) * CC;
  v8f z8 = {0.f, 0.f, 0.f, 0.f, 0.f, 0.f, 0.f, 0.f};
  v8f acc[4] = {z8, z8, z8, z8};
  for (int kk = 0; kk < CC; kk += 32) {
    __builtin_prefetch(arow + kk + 128, 0, 0);
    v16bf af = load_frag(arow + kk + h8, arow + kk + 16 + h8);
    #pragma unroll
    for (int j = 0; j < 4; ++j) {
      const bf16* bp = brow + (size_t)j * 16 * CC + kk;
      v16bf bf = load_frag(bp + h8, bp + 16 + h8);
      acc[j] = wmma_bf16(af, bf, acc[j]);
    }
  }
  #pragma unroll
  for (int j = 0; j < 4; ++j) {
    bf16* zb = Z + (size_t)(mt * 16 + 8 * hh) * CC + ng * 64 + j * 16 + l16;
    #pragma unroll
    for (int r = 0; r < 8; ++r) zb[(size_t)r * CC] = (bf16)acc[j][r];
  }
}

// ---------------- z -> zt [bt][c][544] (pad node) ----------------
__global__ __launch_bounds__(256) void k_transpose_z(const bf16* Z, bf16* Zt) {
  int idx = blockIdx.x * 256 + threadIdx.x;  // exact grid: BT*CC*NP
  int bt = idx / (CC * NP);
  int r  = idx % (CC * NP);
  int c  = r / NP, node = r % NP;
  Zt[idx] = (node < NN) ? Z[((size_t)bt * NN + node) * CC + c] : (bf16)0.0f;
}

// ---------------- h = adj @ z + gcn_b (WMMA, 16x64 per wave) ----------------
__global__ __launch_bounds__(256) void k_gemm_adj(const bf16* adjp, const bf16* Zt,
                                                  const float* bias, bf16* H) {
  int w = threadIdx.x >> 5, lane = threadIdx.x & 31;
  int wt = blockIdx.x * 8 + w;               // 128 * (34*4) wave-tiles, exact
  int bt = wt / 136;
  int rr = wt % 136;
  int mt = rr >> 2, ng = rr & 3;
  int l16 = lane & 15, h8 = (lane >> 4) * 8, hh = lane >> 4;
  const bf16* arow = adjp + (size_t)(mt * 16 + l16) * NP;
  const bf16* brow = Zt + ((size_t)bt * CC + ng * 64 + l16) * NP;
  v8f z8 = {0.f, 0.f, 0.f, 0.f, 0.f, 0.f, 0.f, 0.f};
  v8f acc[4] = {z8, z8, z8, z8};
  for (int kk = 0; kk < NP; kk += 32) {
    v16bf af = load_frag(arow + kk + h8, arow + kk + 16 + h8);
    #pragma unroll
    for (int j = 0; j < 4; ++j) {
      const bf16* bp = brow + (size_t)j * 16 * NP + kk;
      v16bf bf = load_frag(bp + h8, bp + 16 + h8);
      acc[j] = wmma_bf16(af, bf, acc[j]);
    }
  }
  #pragma unroll
  for (int j = 0; j < 4; ++j) {
    int c = ng * 64 + j * 16 + l16;
    float bv = bias[c];
    #pragma unroll
    for (int r = 0; r < 8; ++r) {
      int node = mt * 16 + r + 8 * hh;
      if (node < NN)
        H[((size_t)bt * NN + node) * CC + c] = (bf16)(acc[j][r] + bv);
    }
  }
}

// ---------------- fused temporal conv + BN + ReLU + residual (WMMA) ----------------
// one block per (b, n); stages H[b,:,n,:] with +/-4 halo into LDS (TDM when available)
__global__ __launch_bounds__(256) void k_tcn(const bf16* H, const bf16* Wt /*[K][C][C]*/,
                                             const float* bias, const float* scale,
                                             const float* shift, float* xf, bf16* xb) {
  __shared__ __align__(16) bf16 Hs[TT + 8][CC];   // 72 x 256 bf16 = 36 KB
  int bn = blockIdx.x;
  int b = bn / NN, n = bn % NN;
  int tid = threadIdx.x;

#if __has_builtin(__builtin_amdgcn_tensor_load_to_lds)
  // Tensor Data Mover: DMA rows t=0..63 of H[b,:,n,:] into LDS rows 4..67.
  if (tid < 32) {
    unsigned long long ga =
        (unsigned long long)(const void*)(H + (((size_t)(b * TT)) * NN + n) * CC);
    u32x4 g0;
    g0[0] = 1u;                                   // count=1, user descriptor
    g0[1] = (unsigned)(4 * CC * 2);               // lds_addr = row 4 (bytes)
    g0[2] = (unsigned)(ga & 0xFFFFFFFFu);         // global_addr[31:0]
    g0[3] = (unsigned)((ga >> 32) & 0x01FFFFFFu)  // global_addr[56:32]
            | 0x80000000u;                        // type=2 ("image")
    i32x8 g1;
    g1[0] = 0x00010000;                           // wg_mask=0, data_size=1 (2B)
    g1[1] = (int)(((unsigned)CC & 0xFFFFu) << 16);     // tensor_dim0 = 256 (lo16)
    g1[2] = (int)(((unsigned)TT & 0xFFFFu) << 16);     // dim0 hi=0 | tensor_dim1 = 64 (lo16)
    g1[3] = (int)(((unsigned)CC & 0xFFFFu) << 16);     // dim1 hi=0 | tile_dim0 = 256
    g1[4] = TT;                                   // tile_dim1=64, tile_dim2=0
    g1[5] = NN * CC;                              // tensor_dim0_stride lo32 (elements)
    g1[6] = 0;                                    // stride hi | dim1_stride lo
    g1[7] = 0;
    i32x4 gz = {0, 0, 0, 0};
#if defined(__clang_major__) && (__clang_major__ >= 23)
    i32x8 gz8 = {0, 0, 0, 0, 0, 0, 0, 0};
    __builtin_amdgcn_tensor_load_to_lds(g0, g1, gz, gz, gz8, 0);
#else
    __builtin_amdgcn_tensor_load_to_lds(g0, g1, gz, gz, 0);
#endif
    __builtin_amdgcn_s_wait_tensorcnt(0);
  }
  // zero the halo rows 0..3 and 68..71
  for (int idx = tid; idx < 8 * (CC / 8); idx += 256) {
    int row  = idx >> 5;
    int col8 = (idx & 31) * 8;
    int r2 = (row < 4) ? row : (TT + row);
    uint4 z = {0u, 0u, 0u, 0u};
    *(uint4*)&Hs[r2][col8] = z;
  }
#else
  for (int idx = tid; idx < (TT + 8) * (CC / 8); idx += 256) {
    int row  = idx >> 5;
    int col8 = (idx & 31) * 8;
    int t = row - 4;
    uint4 u = {0u, 0u, 0u, 0u};
    if (t >= 0 && t < TT)
      u = *(const uint4*)&H[(((size_t)(b * TT + t)) * NN + n) * CC + col8];
    *(uint4*)&Hs[row][col8] = u;
  }
#endif
  __syncthreads();

  int w = tid >> 5, lane = tid & 31;
  int l16 = lane & 15, h8 = (lane >> 4) * 8, hh = lane >> 4;
  v8f z8 = {0.f, 0.f, 0.f, 0.f, 0.f, 0.f, 0.f, 0.f};
  for (int ci = 0; ci < 2; ++ci) {
    int ct = w * 2 + ci;                 // 16 cout-tiles across 8 waves
    v8f acc[4] = {z8, z8, z8, z8};       // 4 time-tiles per cout-tile
    #pragma unroll
    for (int k = 0; k < KT; ++k) {
      const bf16* wk = Wt + ((size_t)k * CC + (ct * 16 + l16)) * CC;
      for (int kk = 0; kk < CC; kk += 32) {
        v16bf af = load_frag(wk + kk + h8, wk + kk + 16 + h8);
        #pragma unroll
        for (int tt = 0; tt < 4; ++tt) {
          const bf16* hr = &Hs[tt * 16 + l16 + k][0];
          v16bf bf = load_frag(hr + kk + h8, hr + kk + 16 + h8);
          acc[tt] = wmma_bf16(af, bf, acc[tt]);
        }
      }
    }
    #pragma unroll
    for (int r = 0; r < 8; ++r) {
      int co = ct * 16 + r + 8 * hh;
      float bv = bias[co], sc = scale[co], sh = shift[co];
      #pragma unroll
      for (int tt = 0; tt < 4; ++tt) {
        int t = tt * 16 + l16;
        float y = acc[tt][r] + bv;
        y = y * sc + sh;
        y = fmaxf(y, 0.f);
        size_t off = (((size_t)(b * TT + t)) * NN + n) * CC + co;
        float xn = y + xf[off];
        xf[off] = xn;
        xb[off] = (bf16)xn;
      }
    }
  }
}

// ---------------- head: hid = relu(x @ w1 + b1) (WMMA, 16x64 per wave) ----------------
__global__ __launch_bounds__(256) void k_head1(const bf16* A, const bf16* W1t,
                                               const float* b1, float* hid) {
  int w = threadIdx.x >> 5, lane = threadIdx.x & 31;
  int wt = blockIdx.x * 8 + w;           // (M/16)*2 wave-tiles, exact
  int mt = wt >> 1, ng = wt & 1;
  int l16 = lane & 15, h8 = (lane >> 4) * 8, hh = lane >> 4;
  const bf16* arow = A   + (size_t)(mt * 16 + l16) * CC;
  const bf16* brow = W1t + (size_t)(ng * 64 + l16) * CC;
  v8f z8 = {0.f, 0.f, 0.f, 0.f, 0.f, 0.f, 0.f, 0.f};
  v8f acc[4] = {z8, z8, z8, z8};
  for (int kk = 0; kk < CC; kk += 32) {
    v16bf af = load_frag(arow + kk + h8, arow + kk + 16 + h8);
    #pragma unroll
    for (int j = 0; j < 4; ++j) {
      const bf16* bp = brow + (size_t)j * 16 * CC + kk;
      v16bf bf = load_frag(bp + h8, bp + 16 + h8);
      acc[j] = wmma_bf16(af, bf, acc[j]);
    }
  }
  #pragma unroll
  for (int j = 0; j < 4; ++j) {
    int ncol = ng * 64 + j * 16 + l16;
    float bv = b1[ncol];
    #pragma unroll
    for (int r = 0; r < 8; ++r)
      hid[(size_t)(mt * 16 + r + 8 * hh) * HH2 + ncol] = fmaxf(acc[j][r] + bv, 0.f);
  }
}

// ---------------- head: refined = sigmoid(hid @ w2 + b2) ----------------
__global__ __launch_bounds__(256) void k_head2(const float* hid, const float* w2,
                                               const float* b2, float* refined) {
  int idx = blockIdx.x * 256 + threadIdx.x;
  if (idx >= MROWS * CO) return;
  int row = idx / CO, j = idx % CO;
  const float* hr = hid + (size_t)row * HH2;
  float s = b2[j];
  for (int h = 0; h < HH2; ++h) s += hr[h] * w2[h * CO + j];
  refined[idx] = 1.0f / (1.0f + expf(-s));
}

// ---------------- grouped temporal smoothing + delta ----------------
__global__ __launch_bounds__(256) void k_smooth(const float* refined, const float* sm_w,
                                                const float* sm_b, const float* poses,
                                                float* out) {
  int idx = blockIdx.x * 256 + threadIdx.x;
  const int TOT = BB * TT * NN * CO;
  if (idx >= TOT) return;
  int j = idx % CO;
  int n = (idx / CO) % NN;
  int t = (idx / (CO * NN)) % TT;
  int b = idx / (CO * NN * TT);
  int oc = n * CO + j;
  float s = sm_b[oc];
  #pragma unroll
  for (int jj = 0; jj < CO; ++jj) {
    #pragma unroll
    for (int k = 0; k < SKK; ++k) {
      int t2 = t + k - SKK / 2;
      if (t2 >= 0 && t2 < TT)
        s += refined[(((size_t)(b * TT + t2)) * NN + n) * CO + jj] *
             sm_w[((size_t)oc * CO + jj) * SKK + k];
    }
  }
  out[idx]       = s;              // smoothed
  out[TOT + idx] = s - poses[idx]; // smoothed - poses
}

// ---------------- host ----------------
static inline size_t align_up(size_t x, size_t a) { return (x + a - 1) & ~(a - 1); }

extern "C" void kernel_launch(void* const* d_in, const int* in_sizes, int n_in,
                              void* d_out, int out_size, void* d_ws, size_t ws_size,
                              hipStream_t stream) {
  (void)in_sizes; (void)n_in; (void)out_size; (void)ws_size;
  const float* poses = (const float*)d_in[0];
  const float* adj   = (const float*)d_in[1];
  const float* w_in  = (const float*)d_in[2];
  const float* b_in  = (const float*)d_in[3];
  const float* gcn_w = (const float*)d_in[4];
  const float* gcn_b = (const float*)d_in[5];
  const float* tcn_w = (const float*)d_in[6];
  const float* tcn_b = (const float*)d_in[7];
  const float* bn_g  = (const float*)d_in[8];
  const float* bn_b  = (const float*)d_in[9];
  const float* bn_m  = (const float*)d_in[10];
  const float* bn_v  = (const float*)d_in[11];
  const float* w1    = (const float*)d_in[12];
  const float* b1    = (const float*)d_in[13];
  const float* w2    = (const float*)d_in[14];
  const float* b2    = (const float*)d_in[15];
  const float* sm_w  = (const float*)d_in[16];
  const float* sm_b  = (const float*)d_in[17];
  float* out = (float*)d_out;

  // workspace carve-up
  char* base = (char*)d_ws;
  size_t off = 0;
  auto carve = [&](size_t bytes) { size_t o = off; off = align_up(off + bytes, 256); return base + o; };
  float* xf    = (float*)carve((size_t)MROWS * CC * 4);
  bf16*  xb    = (bf16*) carve((size_t)MROWS * CC * 2);
  bf16*  zb    = (bf16*) carve((size_t)MROWS * CC * 2);   // reused as hid f32 (same bytes)
  bf16*  zt    = (bf16*) carve((size_t)BT * CC * NP * 2); // reused as refined f32
  bf16*  hb    = (bf16*) carve((size_t)MROWS * CC * 2);
  bf16*  gwt   = (bf16*) carve((size_t)LY * CC * CC * 2);
  bf16*  twt   = (bf16*) carve((size_t)LY * KT * CC * CC * 2);
  bf16*  adjp  = (bf16*) carve((size_t)NP * NP * 2);
  bf16*  w1t   = (bf16*) carve((size_t)HH2 * CC * 2);
  float* bnsc  = (float*)carve((size_t)LY * CC * 4);
  float* bnsh  = (float*)carve((size_t)LY * CC * 4);
  float* hid   = (float*)zb;
  float* refined = (float*)zt;

  // weight prep
  k_prep_gcnw<<<(LY * CC * CC + 255) / 256, 256, 0, stream>>>(gcn_w, gwt);
  k_prep_tcnw<<<(LY * KT * CC * CC + 255) / 256, 256, 0, stream>>>(tcn_w, twt);
  k_prep_adj <<<(NP * NP + 255) / 256, 256, 0, stream>>>(adj, adjp);
  k_prep_w1  <<<(HH2 * CC + 255) / 256, 256, 0, stream>>>(w1, w1t);
  k_prep_bn  <<<(LY * CC + 255) / 256, 256, 0, stream>>>(bn_g, bn_b, bn_m, bn_v, bnsc, bnsh);

  // embedding
  k_embed<<<(MROWS * CC) / 256, 256, 0, stream>>>(poses, w_in, b_in, xf, xb);

  // layers
  for (int i = 0; i < LY; ++i) {
    k_gemm_xw<<<(MROWS / 16) * 4 / 8, 256, 0, stream>>>(xb, gwt + (size_t)i * CC * CC, zb);
    k_transpose_z<<<(BT * CC * NP) / 256, 256, 0, stream>>>(zb, zt);
    k_gemm_adj<<<(BT * 136) / 8, 256, 0, stream>>>(adjp, zt, gcn_b + (size_t)i * CC, hb);
    k_tcn<<<BB * NN, 256, 0, stream>>>(hb, twt + (size_t)i * KT * CC * CC,
                                       tcn_b + (size_t)i * CC, bnsc + (size_t)i * CC,
                                       bnsh + (size_t)i * CC, xf, xb);
  }

  // head + smoothing
  k_head1<<<(MROWS / 16) * 2 / 8, 256, 0, stream>>>(xb, w1t, b1, hid);
  k_head2<<<(MROWS * CO + 255) / 256, 256, 0, stream>>>(hid, w2, b2, refined);
  k_smooth<<<(MROWS * CO + 255) / 256, 256, 0, stream>>>(refined, sm_w, sm_b, poses, out);
}
